// LandscapeReconstructor_4329327035090
// MI455X (gfx1250) — matrix operand order repaired and checked by reading, not compiled, not run on hardware
//
#include <hip/hip_runtime.h>
#include <stdint.h>

typedef _Float16 v16h __attribute__((ext_vector_type(16)));
typedef float    v8f  __attribute__((ext_vector_type(8)));

#define BN_EPS 1e-5f

// 16-bit A-matrix 16x32 layout (CDNA5 ISA 7.12.2):
//   lanes 0-15 : M=lane,    dwords 0-3 -> K=0..7,  dwords 4-7 -> K=16..23
//   lanes 16-31: M=lane-16, dwords 0-3 -> K=8..15, dwords 4-7 -> K=24..31
__device__ __forceinline__ int kmapA(int hi, int h) {
    int d = h >> 1, sub = h & 1;
    return ((d >= 4) ? 16 : 0) + (hi ? 8 : 0) + ((d & 3) << 1) + sub;
}

// ---------------------------------------------------------------------------
// Prep: pack pW2/pW3 (fp32 [K][N]) into f16 WMMA B-fragments (lane-contiguous
// 32B per tile -> two b128 loads), and fold BatchNorm into per-channel
// scale/shift so the hot kernels never touch rsqrt.
// ---------------------------------------------------------------------------
__global__ void pack_w_kernel(const float* __restrict__ W2,   // 256x256
                              const float* __restrict__ W3,   // 256x128
                              const float* __restrict__ p_gamma, const float* __restrict__ p_beta,
                              const float* __restrict__ p_mean,  const float* __restrict__ p_var,
                              const float* __restrict__ d_gamma, const float* __restrict__ d_beta,
                              const float* __restrict__ d_mean,  const float* __restrict__ d_var,
                              _Float16* __restrict__ W2pk,
                              _Float16* __restrict__ W3pk,
                              float* __restrict__ bnf) {       // [4*256]
    int idx = blockIdx.x * 256 + threadIdx.x;
    if (idx < 256 * 256) {
        int t = idx >> 9, rem = idx & 511;
        int l = rem >> 4, h = rem & 15;
        int nt = t >> 3, kt = t & 7;
        int K = kt * 32 + ((h >= 8) ? 16 : 0) + ((l >= 16) ? 8 : 0) + (h & 7);   // B 32x16: lanes0-15 K=0..7,16..23
        int N = nt * 16 + (l & 15);
        W2pk[idx] = (_Float16)W2[K * 256 + N];
    } else if (idx < 256 * 256 + 256 * 128) {
        int i = idx - 256 * 256;
        int t = i >> 9, rem = i & 511;
        int l = rem >> 4, h = rem & 15;
        int nt = t >> 3, kt = t & 7;
        int K = kt * 32 + ((h >= 8) ? 16 : 0) + ((l >= 16) ? 8 : 0) + (h & 7);
        int N = nt * 16 + (l & 15);
        W3pk[i] = (_Float16)W3[K * 128 + N];
    } else if (idx < 256 * 256 + 256 * 128 + 256) {
        int c = idx - (256 * 256 + 256 * 128);
        float sp = p_gamma[c] * rsqrtf(p_var[c] + BN_EPS);
        bnf[c]       = sp;                          // s_p
        bnf[256 + c] = p_beta[c] - p_mean[c] * sp;  // t_p
        float sd = d_gamma[c] * rsqrtf(d_var[c] + BN_EPS);
        bnf[512 + c] = sd;                          // s_d
        bnf[768 + c] = d_beta[c] - d_mean[c] * sd;  // t_d
    }
}

// ---------------------------------------------------------------------------
// Main fused kernel: 256 threads = 8 wave32; 32 rows per wave (two 16-row
// A-fragment sets sharing every B tile) -> 256 rows per block.
// LDS: [0,128K) h2 transpose staging, [128K,192K) async-staged W3 tiles.
// ---------------------------------------------------------------------------
__global__ __launch_bounds__(256)
void landscape_kernel(const float* __restrict__ traj,
                      const float* __restrict__ pW1, const float* __restrict__ pb1,
                      const float* __restrict__ pb2, const float* __restrict__ pb3,
                      const float* __restrict__ pW4, const float* __restrict__ pb4,
                      const float* __restrict__ dW1, const float* __restrict__ db1,
                      const float* __restrict__ dW2, const float* __restrict__ db2,
                      const float* __restrict__ fW1, const float* __restrict__ fb1,
                      const float* __restrict__ fW2, const float* __restrict__ fb2,
                      const _Float16* __restrict__ W2pk,
                      const _Float16* __restrict__ W3pk,
                      const float* __restrict__ bnf,
                      float* __restrict__ out, int T) {
    extern __shared__ char smem_raw[];               // 192 KB dynamic LDS
    _Float16* h2pk = (_Float16*)smem_raw;            // 8 waves * 8192 halves
    char*     w3s  = smem_raw + 131072;              // 64 KB W3 tile stage

    const int tid  = threadIdx.x;
    const int wave = tid >> 5;
    const int lane = tid & 31;
    const int lh   = lane & 15;
    const int hi   = lane >> 4;

    // ---- kick off async stage of W3pk tiles into LDS (ASYNCcnt path) ----
    {
        uint32_t lbase = (uint32_t)(uintptr_t)w3s + (uint32_t)tid * 16u;
        uint64_t gbase = (uint64_t)(uintptr_t)W3pk + (uint64_t)tid * 16u;
        #pragma unroll
        for (int i = 0; i < 16; ++i) {               // 256 thr * 16 * 16B = 64 KB
            uint32_t l = lbase + (uint32_t)i * 4096u;
            uint64_t g = gbase + (uint64_t)i * 4096u;
            asm volatile("global_load_async_to_lds_b128 %0, %1, off"
                         :: "v"(l), "v"(g) : "memory");
        }
    }

    const long rowBase  = (long)blockIdx.x * 256;
    const long waveRow0 = rowBase + wave * 32;
    long r0 = waveRow0 + lh;                         // A-rows this lane owns
    long r1 = r0 + 16;
    long rc0 = r0 < T ? r0 : (long)T - 1;            // clamp: EXEC stays all-1s
    long rc1 = r1 < T ? r1 : (long)T - 1;
    float x00 = traj[rc0 * 3 + 0], x01 = traj[rc0 * 3 + 1], x02 = traj[rc0 * 3 + 2];
    float x10 = traj[rc1 * 3 + 0], x11 = traj[rc1 * 3 + 1], x12 = traj[rc1 * 3 + 2];

    // ---- Layer 1 (K=3) + folded BN: A fragments straight into VGPRs ----
    v16h a0[8], a1[8];
    #pragma unroll
    for (int kt = 0; kt < 8; ++kt) {
        #pragma unroll
        for (int h = 0; h < 16; ++h) {
            int k = kt * 32 + kmapA(hi, h);
            float w0 = pW1[k], w1 = pW1[256 + k], w2 = pW1[512 + k];
            float bb = pb1[k], s = bnf[k], t = bnf[256 + k];
            float v0 = fmaf(x00, w0, fmaf(x01, w1, fmaf(x02, w2, bb)));
            float v1 = fmaf(x10, w0, fmaf(x11, w1, fmaf(x12, w2, bb)));
            a0[kt][h] = (_Float16)fmaf(fmaxf(v0, 0.0f), s, t);
            a1[kt][h] = (_Float16)fmaf(fmaxf(v1, 0.0f), s, t);
        }
    }

    // ---- GEMM2: h2 = relu(a1 @ pW2 + pb2); every B tile feeds 2 WMMAs ----
    for (int nt = 0; nt < 16; ++nt) {
        float bias = pb2[nt * 16 + lh];
        v8f c0, c1;
        #pragma unroll
        for (int j = 0; j < 8; ++j) { c0[j] = bias; c1[j] = bias; }
        const v16h* Bp = (const v16h*)(W2pk + (size_t)nt * 8 * 512) + lane;
        #pragma unroll
        for (int kt = 0; kt < 8; ++kt) {
            v16h b = Bp[kt * 32];
            c0 = __builtin_amdgcn_wmma_f32_16x16x32_f16(
                     false, a0[kt], false, b, (short)0, c0, false, false);
            c1 = __builtin_amdgcn_wmma_f32_16x16x32_f16(
                     false, a1[kt], false, b, (short)0, c1, false, false);
        }
        // scatter D (C-layout) -> A-fragment packing in LDS (transpose)
        int n   = nt * 16 + lh;
        int kk  = n & 31;
        int kt2 = n >> 5;
        int hi2 = (kk >> 3) & 1;
        int d   = ((kk & 16) ? 4 : 0) + ((kk & 7) >> 1);
        int h2i = d * 2 + (kk & 1);
        #pragma unroll
        for (int j = 0; j < 8; ++j) {
            int m  = j + hi * 8;
            int lp = m + hi2 * 16;
            int base = wave * 8192 + kt2 * 512 + lp * 16 + h2i;
            h2pk[base]        = (_Float16)fmaxf(c0[j], 0.0f);
            h2pk[base + 4096] = (_Float16)fmaxf(c1[j], 0.0f);
        }
    }

    asm volatile("s_wait_asynccnt 0x0" ::: "memory");  // W3 stage complete (wave)
    __syncthreads();                                   // ...and visible block-wide

    // ---- GEMM3: h3 = relu(h2 @ pW3 + pb3); fold pW4 dot into epilogue ----
    #pragma unroll
    for (int kt = 0; kt < 8; ++kt) {
        a0[kt] = *(const v16h*)(h2pk + wave * 8192 + kt * 512 + lane * 16);
        a1[kt] = *(const v16h*)(h2pk + wave * 8192 + 4096 + kt * 512 + lane * 16);
    }
    float pot0[8], pot1[8];
    #pragma unroll
    for (int j = 0; j < 8; ++j) { pot0[j] = 0.0f; pot1[j] = 0.0f; }

    for (int nt = 0; nt < 8; ++nt) {
        float bias = pb3[nt * 16 + lh];
        v8f c0, c1;
        #pragma unroll
        for (int j = 0; j < 8; ++j) { c0[j] = bias; c1[j] = bias; }
        const v16h* Bp = (const v16h*)(w3s + (size_t)nt * 8192) + lane;  // LDS
        #pragma unroll
        for (int kt = 0; kt < 8; ++kt) {
            v16h b = Bp[kt * 32];
            c0 = __builtin_amdgcn_wmma_f32_16x16x32_f16(
                     false, a0[kt], false, b, (short)0, c0, false, false);
            c1 = __builtin_amdgcn_wmma_f32_16x16x32_f16(
                     false, a1[kt], false, b, (short)0, c1, false, false);
        }
        float w4 = pW4[nt * 16 + lh];
        #pragma unroll
        for (int j = 0; j < 8; ++j) {
            pot0[j] = fmaf(fmaxf(c0[j], 0.0f), w4, pot0[j]);
            pot1[j] = fmaf(fmaxf(c1[j], 0.0f), w4, pot1[j]);
        }
    }
    // reduce partial dots across the 16 lanes sharing each row
    #pragma unroll
    for (int off = 1; off <= 8; off <<= 1) {
        #pragma unroll
        for (int j = 0; j < 8; ++j) {
            pot0[j] += __shfl_xor(pot0[j], off, 32);
            pot1[j] += __shfl_xor(pot1[j], off, 32);
        }
    }
    float b4 = pb4[0];
    #pragma unroll
    for (int j = 0; j < 8; ++j) {
        if (lh == j) {
            long ra = waveRow0 + j + hi * 8;
            long rb = ra + 16;
            if (ra < T) out[ra * 7 + 0] = pot0[j] + b4;
            if (rb < T) out[rb * 7 + 0] = pot1[j] + b4;
        }
    }

    // ---- drift_net + diffusion_net: VALU, exactly 1 thread per row ----
    long row = rowBase + tid;
    long rcd = row < T ? row : (long)T - 1;
    float y0 = traj[rcd * 3 + 0], y1 = traj[rcd * 3 + 1], y2 = traj[rcd * 3 + 2];

    float d0 = 0.f, d1 = 0.f, d2 = 0.f;
    for (int c = 0; c < 256; ++c) {
        float g = fmaf(y0, dW1[c], fmaf(y1, dW1[256 + c], fmaf(y2, dW1[512 + c], db1[c])));
        g = fmaf(fmaxf(g, 0.0f), bnf[512 + c], bnf[768 + c]);   // folded BN
        d0 = fmaf(g, dW2[c * 3 + 0], d0);
        d1 = fmaf(g, dW2[c * 3 + 1], d1);
        d2 = fmaf(g, dW2[c * 3 + 2], d2);
    }
    float f0 = 0.f, f1 = 0.f, f2 = 0.f;
    for (int c = 0; c < 128; ++c) {
        float fv = fmaf(y0, fW1[c], fmaf(y1, fW1[128 + c], fmaf(y2, fW1[256 + c], fb1[c])));
        fv = fmaxf(fv, 0.0f);
        f0 = fmaf(fv, fW2[c * 3 + 0], f0);
        f1 = fmaf(fv, fW2[c * 3 + 1], f1);
        f2 = fmaf(fv, fW2[c * 3 + 2], f2);
    }
    if (row < T) {
        out[row * 7 + 1] = d0 + db2[0];
        out[row * 7 + 2] = d1 + db2[1];
        out[row * 7 + 3] = d2 + db2[2];
        float z0 = f0 + fb2[0], z1 = f1 + fb2[1], z2 = f2 + fb2[2];
        out[row * 7 + 4] = (z0 > 20.0f) ? z0 : log1pf(__expf(z0));
        out[row * 7 + 5] = (z1 > 20.0f) ? z1 : log1pf(__expf(z1));
        out[row * 7 + 6] = (z2 > 20.0f) ? z2 : log1pf(__expf(z2));
    }
}

extern "C" void kernel_launch(void* const* d_in, const int* in_sizes, int n_in,
                              void* d_out, int out_size, void* d_ws, size_t ws_size,
                              hipStream_t stream) {
    const float* traj    = (const float*)d_in[0];
    const float* pW1     = (const float*)d_in[1];
    const float* pb1     = (const float*)d_in[2];
    const float* p_gamma = (const float*)d_in[3];
    const float* p_beta  = (const float*)d_in[4];
    const float* p_mean  = (const float*)d_in[5];
    const float* p_var   = (const float*)d_in[6];
    const float* pW2     = (const float*)d_in[7];
    const float* pb2     = (const float*)d_in[8];
    const float* pW3     = (const float*)d_in[9];
    const float* pb3     = (const float*)d_in[10];
    const float* pW4     = (const float*)d_in[11];
    const float* pb4     = (const float*)d_in[12];
    const float* dW1     = (const float*)d_in[13];
    const float* db1     = (const float*)d_in[14];
    const float* d_gamma = (const float*)d_in[15];
    const float* d_beta  = (const float*)d_in[16];
    const float* d_mean  = (const float*)d_in[17];
    const float* d_var   = (const float*)d_in[18];
    const float* dW2     = (const float*)d_in[19];
    const float* db2     = (const float*)d_in[20];
    const float* fW1     = (const float*)d_in[21];
    const float* fb1     = (const float*)d_in[22];
    const float* fW2     = (const float*)d_in[23];
    const float* fb2     = (const float*)d_in[24];

    const int T = in_sizes[0] / 3;

    // workspace: W2pk 128 KB | W3pk 64 KB | bnf 4 KB
    _Float16* W2pk = (_Float16*)d_ws;
    _Float16* W3pk = (_Float16*)((char*)d_ws + 131072);
    float*    bnf  = (float*)((char*)d_ws + 196608);

    const int packElems = 256 * 256 + 256 * 128 + 256;
    pack_w_kernel<<<(packElems + 255) / 256, 256, 0, stream>>>(
        pW2, pW3,
        p_gamma, p_beta, p_mean, p_var,
        d_gamma, d_beta, d_mean, d_var,
        W2pk, W3pk, bnf);

    const int nblocks = (T + 255) / 256;
    landscape_kernel<<<nblocks, 256, 196608, stream>>>(
        traj, pW1, pb1, pb2, pb3, pW4, pb4,
        dW1, db1, dW2, db2,
        fW1, fb1, fW2, fb2,
        W2pk, W3pk, bnf,
        (float*)d_out, T);
}
